// QalbAttention_78975858638965
// MI455X (gfx1250) — matrix operand order repaired and checked by reading, not compiled, not run on hardware
//
#include <hip/hip_runtime.h>
#include <hip/hip_bf16.h>

#define Bb 2
#define Nn 2048
#define Dd 1024
#define Hh 16
#define HD 64

typedef __attribute__((ext_vector_type(16))) _Float16 v16h;
typedef __attribute__((ext_vector_type(8)))  _Float16 v8h;
typedef __attribute__((ext_vector_type(8)))  float    v8f;
typedef unsigned int v4u __attribute__((ext_vector_type(4)));
typedef int v8i __attribute__((ext_vector_type(8)));
typedef int v4i __attribute__((ext_vector_type(4)));

#define WMMA_F16(A, Bv, C) \
  __builtin_amdgcn_wmma_f32_16x16x32_f16(false, (A), false, (Bv), (short)0, (C), false, false)

// Build a 16-half fragment from two 16-byte chunks (works for global and LDS).
static __device__ __forceinline__ v16h ld_frag16(const _Float16* p0, const _Float16* p1) {
  v8h a = *(const v8h*)p0;
  v8h b = *(const v8h*)p1;
  return __builtin_shufflevector(a, b, 0,1,2,3,4,5,6,7,8,9,10,11,12,13,14,15);
}

// ---------------------------------------------------------------------------
// Tensor Data Mover: load a rows x cols (f16) 2-D tile from global into LDS.
// D# per CDNA5 ISA 8.3/8.4: group0 {count=1, lds_addr, global_addr, type=2},
// group1 {data_size=2B, tensor_dim0=cols, tensor_dim1=rows, tile_dim0=cols,
// tile_dim1=rows, tensor_dim0_stride=row_stride}; groups 2/3 zero (2-D).
// ---------------------------------------------------------------------------
static __device__ __forceinline__ void tdm_load_tile_2d(
    unsigned lds_off, const _Float16* gsrc,
    unsigned rows, unsigned cols, unsigned row_stride_elts) {
  unsigned long long ga = (unsigned long long)(uintptr_t)gsrc;
  v4u g0 = { 1u,                                            // count=1 (user D#)
             lds_off,                                       // LDS byte address
             (unsigned)ga,                                  // global_addr[31:0]
             (unsigned)((ga >> 32) & 0x1FFFFFFu) | (2u << 30) }; // [56:32]|type=2
  v8i g1;
  g1[0] = (int)(1u << 16);                                  // data_size = 2 bytes
  g1[1] = (int)((cols & 0xFFFFu) << 16);                    // tensor_dim0 lo16
  g1[2] = (int)((cols >> 16) | ((rows & 0xFFFFu) << 16));   // td0 hi | td1 lo
  g1[3] = (int)((rows >> 16) | (cols << 16));               // td1 hi | tile_dim0
  g1[4] = (int)rows;                                        // tile_dim1 (dim2=0)
  g1[5] = (int)row_stride_elts;                             // dim0_stride[31:0]
  g1[6] = 0;
  g1[7] = 0;
  v4i z4 = {0, 0, 0, 0};
#if defined(__clang_major__) && __clang_major__ >= 23
  v8i z8 = {0, 0, 0, 0, 0, 0, 0, 0};
  __builtin_amdgcn_tensor_load_to_lds(g0, g1, z4, z4, z8, 0);
#else
  __builtin_amdgcn_tensor_load_to_lds(g0, g1, z4, z4, 0);
#endif
}

// ---------------------------------------------------------------------------
// Per-batch temperature: rscale[b] = 1 / (sqrt(HD) * psi_b)
// ---------------------------------------------------------------------------
__global__ __launch_bounds__(256)
void scale_kernel(const float* __restrict__ x, const float* __restrict__ cw,
                  const float* __restrict__ cb, const float* __restrict__ alpha,
                  const float* __restrict__ Tb, const int* __restrict__ tstep,
                  float* __restrict__ rscale) {
  int b = blockIdx.x, tid = threadIdx.x;
  float part = 0.f;
  for (int n = 0; n < Nn; ++n) {
    const float* row = x + ((size_t)b * Nn + n) * Dd;
    for (int d = tid; d < Dd; d += 256) part += row[d] * cw[d];
  }
  part *= (1.0f / Nn);
  __shared__ float red[256];
  red[tid] = part; __syncthreads();
  for (int s = 128; s > 0; s >>= 1) {
    if (tid < s) red[tid] += red[tid + s];
    __syncthreads();
  }
  if (tid == 0) {
    float z = red[0] + cb[0];
    float cx = 1.f / (1.f + expf(-z));
    float Teff = (fabsf(Tb[0]) + 1.f) * (0.5f + cx);
    float psi = 1.f + alpha[0] * sinf(6.28318530718f * (float)tstep[0] / Teff);
    rscale[b] = 1.f / (8.f * psi);   // sqrt(64) = 8
  }
}

// ---------------------------------------------------------------------------
// fp32 -> f16 conversion
// ---------------------------------------------------------------------------
__global__ __launch_bounds__(256)
void cvt_f32_f16(const float* __restrict__ s, _Float16* __restrict__ d, int n) {
  int i = blockIdx.x * 256 + threadIdx.x;
  if (i < n) d[i] = (_Float16)s[i];
}

// ---------------------------------------------------------------------------
// C[m,e] = sum_k A[m,k] * W[e,k]   (A: MxK f16 row-major, W: NoutxK f16 row-major)
// Block = 8 waves sharing one 64-row A block (Nout/64 must be a multiple of 8
// per row block -- true here: Nout/64 = 16). The A K-slice (64x32 f16, 4 KB)
// is staged into LDS by the Tensor Data Mover, double-buffered so the DMA
// overlaps the 16 WMMAs of the current slice; waves read A fragments from LDS
// (ds_load_b128) and B fragments from global (weights are unique per wave).
// Grid must exactly cover (M/64)*(Nout/64)/8 blocks (no divergent barriers).
// ---------------------------------------------------------------------------
__global__ __launch_bounds__(256)
void gemm_nt_f16f32(const _Float16* __restrict__ A, const _Float16* __restrict__ W,
                    float* __restrict__ C, int M, int Nout, int K) {
  __shared__ __align__(16) _Float16 atile[2][64 * 32];
  int lane = threadIdx.x & 31;
  int wave = threadIdx.x >> 5;
  int ntile = Nout >> 6;                       // 64-wide column tiles
  long tile = (long)blockIdx.x * 8 + wave;
  int rt = (int)(tile / ntile), ct = (int)(tile % ntile);   // rt uniform in block
  int ln = lane & 15, hlf = lane >> 4;

  // Flat shared pointers carry the LDS byte offset in their low 32 bits.
  unsigned lds0 = (unsigned)(uintptr_t)&atile[0][0];
  unsigned lds1 = (unsigned)(uintptr_t)&atile[1][0];

  const _Float16* abase = A + (size_t)(rt * 64) * K;
  const _Float16* brow  = W + (size_t)(ct * 64 + ln) * K + (hlf << 4);

  v8f acc[4][4];
  #pragma unroll
  for (int i = 0; i < 4; ++i)
    #pragma unroll
    for (int j = 0; j < 4; ++j) acc[i][j] = (v8f){};

  // Prologue: stage the first A slice.
  if (wave == 0) {
    tdm_load_tile_2d(lds0, abase, 64, 32, (unsigned)K);
    __builtin_amdgcn_s_wait_tensorcnt(0);
  }
  __syncthreads();

  int buf = 0;
  for (int kk = 0; kk < K; kk += 32) {
    // Issue the TDM for the next slice into the other buffer (overlaps WMMA).
    if (wave == 0 && kk + 32 < K)
      tdm_load_tile_2d(buf ? lds0 : lds1, abase + kk + 32, 64, 32, (unsigned)K);

    __builtin_prefetch(brow + kk + 256, 0, 1);          // -> global_prefetch_b8

    const _Float16* ab = &atile[buf][0];
    v16h a[4];
    #pragma unroll
    for (int i = 0; i < 4; ++i) {
      const _Float16* p = ab + (unsigned)(i * 16 + ln) * 32;
      a[i] = ld_frag16(p + hlf * 16, p + hlf * 16 + 8);  // ds_load_b128 x2
    }
    #pragma unroll
    for (int j = 0; j < 4; ++j) {
      const _Float16* p = brow + (size_t)(j * 16) * K + kk;
      v16h bfr = ld_frag16(p, p + 8);
      #pragma unroll
      for (int i = 0; i < 4; ++i)
        acc[i][j] = WMMA_F16(a[i], bfr, acc[i][j]);
    }

    if (wave == 0 && kk + 32 < K) __builtin_amdgcn_s_wait_tensorcnt(0);
    __syncthreads();                                    // slice ready / reads done
    buf ^= 1;
  }

  // C-layout: vgpr r -> row (8*hlf + r), lane ln -> column
  #pragma unroll
  for (int i = 0; i < 4; ++i) {
    float* crow = C + (size_t)(rt * 64 + i * 16 + hlf * 8) * Nout + ct * 64 + ln;
    #pragma unroll
    for (int j = 0; j < 4; ++j)
      #pragma unroll
      for (int r = 0; r < 8; ++r)
        crow[(size_t)r * Nout + j * 16] = acc[i][j][r];
  }
}

// ---------------------------------------------------------------------------
// RoPE on q,k; convert to f16 in (B,H,N,HD); V converted transposed (B,H,HD,N)
// so the flash-attention B-fragment for P*V is two contiguous b128 loads.
// ---------------------------------------------------------------------------
__global__ __launch_bounds__(256)
void rope_cvt(const float* __restrict__ qf, const float* __restrict__ kf,
              const float* __restrict__ vf, _Float16* __restrict__ Qh,
              _Float16* __restrict__ Kh, _Float16* __restrict__ Vt) {
  int idx = blockIdx.x * 256 + threadIdx.x;      // (b,h,n,hd)
  int hd = idx & 63;
  int n  = (idx >> 6) & (Nn - 1);
  int h  = (idx >> 17) & (Hh - 1);
  int b  = idx >> 21;
  size_t src = ((size_t)(b * Nn + n)) * Dd + h * HD + hd;
  int i = hd & 31;
  float inv = exp2f(-(float)(2 * i) * (13.2877123795f / 64.0f)); // base 10000
  float ang = (float)n * inv;
  float c = cosf(ang), s = sinf(ang);
  float qv = qf[src], kv = kf[src];
  float qr, kr;
  if (hd < 32) { qr = -qf[src + 32]; kr = -kf[src + 32]; }
  else         { qr =  qf[src - 32]; kr =  kf[src - 32]; }
  Qh[idx] = (_Float16)(qv * c + qr * s);
  Kh[idx] = (_Float16)(kv * c + kr * s);
  Vt[(((size_t)(b * Hh + h)) * HD + hd) * Nn + n] = (_Float16)vf[src];
}

// ---------------------------------------------------------------------------
// Causal flash attention. One wave per (b,h,16-query tile), 32 keys per step.
// S = Q K^T (4 WMMA), online softmax with wave32 shfl_xor reductions,
// P relayout via per-wave LDS, O += P V (4 WMMA).
// ---------------------------------------------------------------------------
__global__ __launch_bounds__(256)
void flash_attn(const _Float16* __restrict__ Qh, const _Float16* __restrict__ Kh,
                const _Float16* __restrict__ Vt, const float* __restrict__ rscale,
                _Float16* __restrict__ Oh) {
  __shared__ __align__(16) _Float16 pbuf[8][16][32];
  int lane = threadIdx.x & 31, w = threadIdx.x >> 5;
  int wid = blockIdx.x * 8 + w;
  int qt = wid & 127;            // N/16 = 128
  int h  = (wid >> 7) & 15;
  int b  = wid >> 11;
  int q0 = qt * 16;
  int hlf = lane >> 4, ln = lane & 15;
  float rs = rscale[b];
  const size_t bh = (size_t)(b * Hh + h);

  // Preload Q fragments (K-dim 0..31 and 32..63)
  const _Float16* qrow = Qh + (bh * Nn + q0 + ln) * HD;
  v16h aq0 = ld_frag16(qrow + hlf * 16,      qrow + hlf * 16 + 8);
  v16h aq1 = ld_frag16(qrow + 32 + hlf * 16, qrow + 40 + hlf * 16);

  v8f o0 = {}, o1 = {}, o2 = {}, o3 = {};
  float mi[8], li[8];
  #pragma unroll
  for (int r = 0; r < 8; ++r) { mi[r] = -3.0e38f; li[r] = 0.f; }

  int nsteps = (q0 + 15) / 32 + 1;
  for (int st = 0; st < nsteps; ++st) {
    int jb = st * 32;
    // ---- S = Q K^T for keys [jb, jb+32) ----
    const _Float16* k0 = Kh + (bh * Nn + jb + ln) * HD + (hlf << 4);
    const _Float16* k1 = k0 + 16 * HD;
    v8f s0 = {}, s1 = {};
    v16h bk;
    bk = ld_frag16(k0,      k0 + 8);  s0 = WMMA_F16(aq0, bk, s0);
    bk = ld_frag16(k0 + 32, k0 + 40); s0 = WMMA_F16(aq1, bk, s0);
    bk = ld_frag16(k1,      k1 + 8);  s1 = WMMA_F16(aq0, bk, s1);
    bk = ld_frag16(k1 + 32, k1 + 40); s1 = WMMA_F16(aq1, bk, s1);

    // ---- scale, causal mask, online softmax ----
    float p0[8], p1[8], corr[8];
    #pragma unroll
    for (int r = 0; r < 8; ++r) {
      int qr_i = q0 + r + 8 * hlf;
      float v0 = s0[r] * rs; if (jb + ln      > qr_i) v0 = -3.0e38f;
      float v1 = s1[r] * rs; if (jb + 16 + ln > qr_i) v1 = -3.0e38f;
      float rm = fmaxf(v0, v1);
      rm = fmaxf(rm, __shfl_xor(rm, 1, 32));
      rm = fmaxf(rm, __shfl_xor(rm, 2, 32));
      rm = fmaxf(rm, __shfl_xor(rm, 4, 32));
      rm = fmaxf(rm, __shfl_xor(rm, 8, 32));
      float mnew = fmaxf(mi[r], rm);
      float cr = expf(mi[r] - mnew);
      float e0 = expf(v0 - mnew), e1 = expf(v1 - mnew);
      float rsum = e0 + e1;
      rsum += __shfl_xor(rsum, 1, 32);
      rsum += __shfl_xor(rsum, 2, 32);
      rsum += __shfl_xor(rsum, 4, 32);
      rsum += __shfl_xor(rsum, 8, 32);
      li[r] = li[r] * cr + rsum;
      mi[r] = mnew;
      corr[r] = cr;
      p0[r] = e0; p1[r] = e1;
    }
    #pragma unroll
    for (int r = 0; r < 8; ++r) {
      o0[r] *= corr[r]; o1[r] *= corr[r]; o2[r] *= corr[r]; o3[r] *= corr[r];
    }

    // ---- relayout P (C-layout) -> A-fragment via per-wave LDS ----
    #pragma unroll
    for (int r = 0; r < 8; ++r) {
      pbuf[w][r + 8 * hlf][ln]      = (_Float16)p0[r];
      pbuf[w][r + 8 * hlf][16 + ln] = (_Float16)p1[r];
    }
    asm volatile("" ::: "memory");   // same-wave LDS is in-order (DScnt)
    const _Float16* pr = &pbuf[w][ln][0];
    v16h ap = ld_frag16(pr + hlf * 16, pr + hlf * 16 + 8);

    // ---- O += P * V ; V is (B,H,HD,N) so each lane reads contiguous keys ----
    const _Float16* vb = Vt + (bh * HD) * Nn + jb + (hlf << 4);
    v16h bv;
    bv = ld_frag16(vb + (size_t)(ln)      * Nn, vb + (size_t)(ln)      * Nn + 8);
    o0 = WMMA_F16(ap, bv, o0);
    bv = ld_frag16(vb + (size_t)(16 + ln) * Nn, vb + (size_t)(16 + ln) * Nn + 8);
    o1 = WMMA_F16(ap, bv, o1);
    bv = ld_frag16(vb + (size_t)(32 + ln) * Nn, vb + (size_t)(32 + ln) * Nn + 8);
    o2 = WMMA_F16(ap, bv, o2);
    bv = ld_frag16(vb + (size_t)(48 + ln) * Nn, vb + (size_t)(48 + ln) * Nn + 8);
    o3 = WMMA_F16(ap, bv, o3);
  }

  // ---- normalize and store to (B,N,D) f16 for the Wo GEMM ----
  float il[8];
  #pragma unroll
  for (int r = 0; r < 8; ++r) il[r] = 1.f / li[r];
  _Float16* obase = Oh + ((size_t)b * Nn + q0 + 8 * hlf) * Dd + h * HD + ln;
  #pragma unroll
  for (int r = 0; r < 8; ++r) {
    _Float16* op = obase + (size_t)r * Dd;
    op[0]  = (_Float16)(o0[r] * il[r]);
    op[16] = (_Float16)(o1[r] * il[r]);
    op[32] = (_Float16)(o2[r] * il[r]);
    op[48] = (_Float16)(o3[r] * il[r]);
  }
}

// ---------------------------------------------------------------------------
extern "C" void kernel_launch(void* const* d_in, const int* in_sizes, int n_in,
                              void* d_out, int out_size, void* d_ws, size_t ws_size,
                              hipStream_t stream) {
  (void)in_sizes; (void)n_in; (void)out_size; (void)ws_size;
  const float* x     = (const float*)d_in[0];
  const float* Wq    = (const float*)d_in[1];
  const float* Wk    = (const float*)d_in[2];
  const float* Wv    = (const float*)d_in[3];
  const float* Wo    = (const float*)d_in[4];
  const float* alpha = (const float*)d_in[5];
  const float* Tb    = (const float*)d_in[6];
  const float* cw    = (const float*)d_in[7];
  const float* cb    = (const float*)d_in[8];
  const int*   tstep = (const int*)d_in[11];
  float* out = (float*)d_out;

  char* ws = (char*)d_ws;
  const size_t MB = 1024 * 1024;
  _Float16* xh  = (_Float16*)(ws + 0);        // 8 MB
  _Float16* wqh = (_Float16*)(ws + 8  * MB);  // 2 MB each
  _Float16* wkh = (_Float16*)(ws + 10 * MB);
  _Float16* wvh = (_Float16*)(ws + 12 * MB);
  _Float16* woh = (_Float16*)(ws + 14 * MB);
  float*    qf  = (float*)   (ws + 16 * MB);  // 16 MB each
  float*    kf  = (float*)   (ws + 32 * MB);
  float*    vf  = (float*)   (ws + 48 * MB);
  _Float16* Qh  = (_Float16*)(ws + 64 * MB);  // 8 MB each
  _Float16* Kh  = (_Float16*)(ws + 72 * MB);
  _Float16* Vt  = (_Float16*)(ws + 80 * MB);
  _Float16* Oh  = (_Float16*)(ws + 88 * MB);
  float* rscale = (float*)   (ws + 96 * MB);

  scale_kernel<<<Bb, 256, 0, stream>>>(x, cw, cb, alpha, Tb, tstep, rscale);

  cvt_f32_f16<<<(Bb * Nn * Dd) / 256, 256, 0, stream>>>(x,  xh,  Bb * Nn * Dd);
  cvt_f32_f16<<<(Dd * Dd) / 256, 256, 0, stream>>>(Wq, wqh, Dd * Dd);
  cvt_f32_f16<<<(Dd * Dd) / 256, 256, 0, stream>>>(Wk, wkh, Dd * Dd);
  cvt_f32_f16<<<(Dd * Dd) / 256, 256, 0, stream>>>(Wv, wvh, Dd * Dd);
  cvt_f32_f16<<<(Dd * Dd) / 256, 256, 0, stream>>>(Wo, woh, Dd * Dd);

  int tiles64 = (Bb * Nn / 64) * (Dd / 64);   // 1024 wave-tiles, 8 waves/block
  gemm_nt_f16f32<<<tiles64 / 8, 256, 0, stream>>>(xh, wqh, qf, Bb * Nn, Dd, Dd);
  gemm_nt_f16f32<<<tiles64 / 8, 256, 0, stream>>>(xh, wkh, kf, Bb * Nn, Dd, Dd);
  gemm_nt_f16f32<<<tiles64 / 8, 256, 0, stream>>>(xh, wvh, vf, Bb * Nn, Dd, Dd);

  rope_cvt<<<(Bb * Hh * Nn * HD) / 256, 256, 0, stream>>>(qf, kf, vf, Qh, Kh, Vt);

  flash_attn<<<(Bb * Hh * (Nn / 16)) / 8, 256, 0, stream>>>(Qh, Kh, Vt, rscale, Oh);

  gemm_nt_f16f32<<<tiles64 / 8, 256, 0, stream>>>(Oh, woh, out, Bb * Nn, Dd, Dd);
}
